// ABMIL_11776800326091
// MI455X (gfx1250) — matrix operand order repaired and checked by reading, not compiled, not run on hardware
//
#include <hip/hip_runtime.h>

#define B_   4
#define N_   20000
#define IN_  1024
#define EMB_ 512
#define ATT_ 128
#define MT_  64            // rows per workgroup
#define NT1_ 313           // ceil(20000/64)

typedef __attribute__((ext_vector_type(16))) __bf16 v16bf;
typedef __attribute__((ext_vector_type(8)))  float  v8f;

union FragU { uint4 q[2]; v16bf v; };

// ---------- workspace layout (bytes, 256-aligned) ----------
static constexpr size_t OFF_WPEP = 0;                               // 1,048,576
static constexpr size_t OFF_WVP  = OFF_WPEP + (size_t)IN_*EMB_*2;
static constexpr size_t OFF_WUP  = OFF_WVP  + (size_t)EMB_*ATT_*2;
static constexpr size_t OFF_H    = OFF_WUP  + (size_t)EMB_*ATT_*2;
static constexpr size_t OFF_A    = OFF_H    + (size_t)B_*N_*EMB_*2; // +81,920,000
static constexpr size_t OFF_ST   = OFF_A    + (size_t)B_*N_*4;      // +320,000
static constexpr size_t OFF_M    = OFF_ST   + 256;

// ---------- helpers ----------
__device__ __forceinline__ unsigned f2bf(float f) {
  unsigned u = __float_as_uint(f);
  u += 0x7FFFu + ((u >> 16) & 1u);   // round-to-nearest-even
  return u >> 16;
}
__device__ __forceinline__ float bf2f(unsigned short h) {
  return __uint_as_float(((unsigned)h) << 16);
}

// A fragment (16-bit A 16x32): lane row M=lane&15; dwords [o..o+3],[o+8..o+11]; o=4*(lane>=16)
__device__ __forceinline__ v16bf load_a_frag(const unsigned short* row, int kt, int odw) {
  const uint4* p = (const uint4*)(row + kt * 32 + (odw << 1));
  FragU u; u.q[0] = p[0]; u.q[1] = p[2];
  return u.v;
}
// B fragment pre-packed: frag = 32 lanes x 8 dwords, lane reads 32B contiguously
__device__ __forceinline__ v16bf load_b_frag(const unsigned* W, int frag, int lane) {
  const uint4* p = (const uint4*)(W + ((size_t)frag << 8) + (lane << 3));
  FragU u; u.q[0] = p[0]; u.q[1] = p[1];
  return u.v;
}
__device__ __forceinline__ v8f wmma_bf16(v16bf a, v16bf b, v8f c) {
  return __builtin_amdgcn_wmma_f32_16x16x32_bf16(false, a, false, b, (short)0, c, false, false);
}

// ---------- kernel 0: pack weights fp32 -> bf16 B-fragment layout; zero M ----------
// B 32x16 16-bit layout: lane = col n | khi; k0 = kt*32 + khi*8 + 2*(v&3) + 16*(v>>2)
__device__ __forceinline__ void pack_one(const float* W, unsigned* dst, int d, int KT, int Nn) {
  int frag   = d >> 8;
  int within = d & 255;
  int lane   = within >> 3;
  int v      = within & 7;
  int nt  = frag / KT;
  int kt  = frag - nt * KT;
  int n   = nt * 16 + (lane & 15);
  int khi = (lane >> 4) & 1;
  int k0  = kt * 32 + khi * 8 + 2 * (v & 3) + ((v >> 2) << 4);
  unsigned lo = f2bf(W[(size_t)k0 * Nn + n]);
  unsigned hi = f2bf(W[((size_t)k0 + 1) * Nn + n]);
  dst[d] = lo | (hi << 16);
}

__global__ __launch_bounds__(256) void k_prep(const float* __restrict__ Wpe,
                                              const float* __restrict__ Wv,
                                              const float* __restrict__ Wu,
                                              unsigned* __restrict__ WPEP,
                                              unsigned* __restrict__ WVP,
                                              unsigned* __restrict__ WUP,
                                              float* __restrict__ M) {
  int gid = blockIdx.x * 256 + threadIdx.x;
  const int D1 = IN_ * EMB_ / 2;
  const int D2 = EMB_ * ATT_ / 2;
  if (gid < D1)                pack_one(Wpe, WPEP, gid,            IN_ / 32, EMB_);
  else if (gid < D1 + D2)      pack_one(Wv,  WVP,  gid - D1,       EMB_ / 32, ATT_);
  else if (gid < D1 + 2 * D2)  pack_one(Wu,  WUP,  gid - D1 - D2,  EMB_ / 32, ATT_);
  else {
    int z = gid - D1 - 2 * D2;
    if (z < B_ * EMB_) M[z] = 0.0f;
  }
}

// ---------- kernel 1: fused patch-embed GEMM + gated attention (64-row tiles) ----------
__global__ __launch_bounds__(256) void k_main(
    const float* __restrict__ x, const float* __restrict__ b_pe,
    const float* __restrict__ b_V, const float* __restrict__ b_U,
    const float* __restrict__ w_att, const float* __restrict__ b_att,
    const unsigned* __restrict__ WPEP, const unsigned* __restrict__ WVP,
    const unsigned* __restrict__ WUP,
    unsigned short* __restrict__ H, float* __restrict__ Alog) {
  constexpr int HS = EMB_ + 8;   // 520 bf16 stride (4-bank skew, 16B-aligned rows)
  // One buffer, reused: first as x K-chunk [64 x 512] bf16, then as h tile [64 x 512] bf16
  __shared__ __align__(16) unsigned short lds_buf[MT_ * HS];   // 66,560 B
  __shared__ float lds_a[MT_];

  const int tile = blockIdx.x;
  const int b    = blockIdx.y;
  const int n0   = tile * MT_;
  const int t    = threadIdx.x;
  const int lane = t & 31;
  const int w    = t >> 5;
  const int m    = lane & 15;
  const int hi16 = (lane >> 4) & 1;
  const int odw  = hi16 << 2;

  if (t < MT_) lds_a[t] = 0.0f;

  v8f acc[4][4] = {};   // [row sub-tile][n-tile]

  // ---- GEMM1 over two K chunks of 512, staged through lds_buf ----
  for (int kc = 0; kc < 2; ++kc) {
    // stage x[:, kc*512 .. +512) for 64 rows (clamped) -> bf16 LDS
    #pragma unroll 4
    for (int i = 0; i < 32; ++i) {
      int f   = t + (i << 8);          // float4 index, 64 rows x 128 f4/row
      int row = f >> 7;
      int c4  = f & 127;
      int rg  = n0 + row; if (rg >= N_) rg = N_ - 1;
      float4 v = ((const float4*)(x + ((size_t)b * N_ + rg) * IN_ + kc * (IN_/2)))[c4];
      uint2 pk;
      pk.x = f2bf(v.x) | (f2bf(v.y) << 16);
      pk.y = f2bf(v.z) | (f2bf(v.w) << 16);
      *(uint2*)&lds_buf[row * HS + (c4 << 2)] = pk;
    }
    __syncthreads();

    #pragma unroll 1
    for (int kt = 0; kt < 16; ++kt) {
      v16bf af[4];
      #pragma unroll
      for (int s = 0; s < 4; ++s)
        af[s] = load_a_frag(lds_buf + (s * 16 + m) * HS, kt, odw);
      int ktg = kc * 16 + kt;
      #pragma unroll
      for (int nt = 0; nt < 4; ++nt) {
        v16bf bf = load_b_frag(WPEP, (w * 4 + nt) * (IN_ / 32) + ktg, lane);
        #pragma unroll
        for (int s = 0; s < 4; ++s)
          acc[s][nt] = wmma_bf16(af[s], bf, acc[s][nt]);
      }
    }
    __syncthreads();   // everyone done reading before buffer is overwritten
  }

  // ---- bias + park h (bf16) in lds_buf (now the h tile) ----
  #pragma unroll
  for (int nt = 0; nt < 4; ++nt) {
    int col = w * 64 + nt * 16 + m;
    float bias = b_pe[col];
    #pragma unroll
    for (int s = 0; s < 4; ++s) {
      #pragma unroll
      for (int r = 0; r < 8; ++r) {
        lds_buf[(s * 16 + r + (hi16 << 3)) * HS + col] =
            (unsigned short)f2bf(acc[s][nt][r] + bias);
      }
    }
  }
  __syncthreads();

  // ---- stream h tile to global (bf16), guarded rows ----
  unsigned short* Hg = H + ((size_t)b * N_ + n0) * EMB_;
  #pragma unroll 4
  for (int i = 0; i < 16; ++i) {
    int f   = t + (i << 8);            // uint4 index: 64 rows x 64 per row
    int row = f >> 6;
    int c   = f & 63;
    if (n0 + row < N_)
      *(uint4*)&Hg[(size_t)row * EMB_ + (c << 3)] = *(const uint4*)&lds_buf[row * HS + (c << 3)];
  }

  // ---- GEMM2: A_V/A_U (64x128, K=512); wave w owns 16 att cols ----
  v8f accv[4] = {}, accu[4] = {};
  #pragma unroll 1
  for (int kt = 0; kt < EMB_ / 32; ++kt) {
    v16bf af[4];
    #pragma unroll
    for (int s = 0; s < 4; ++s)
      af[s] = load_a_frag(lds_buf + (s * 16 + m) * HS, kt, odw);
    v16bf bv = load_b_frag(WVP, w * (EMB_ / 32) + kt, lane);
    v16bf bu = load_b_frag(WUP, w * (EMB_ / 32) + kt, lane);
    #pragma unroll
    for (int s = 0; s < 4; ++s) {
      accv[s] = wmma_bf16(af[s], bv, accv[s]);
      accu[s] = wmma_bf16(af[s], bu, accu[s]);
    }
  }
  {
    int col = w * 16 + m;
    float bV = b_V[col], bU = b_U[col], wa = w_att[col];
    #pragma unroll
    for (int s = 0; s < 4; ++s) {
      #pragma unroll
      for (int r = 0; r < 8; ++r) {
        float gv = tanhf(accv[s][r] + bV);
        float gu = 1.0f / (1.0f + expf(-(accu[s][r] + bU)));
        float g  = gv * gu * wa;
        g += __shfl_xor(g, 1);
        g += __shfl_xor(g, 2);
        g += __shfl_xor(g, 4);
        g += __shfl_xor(g, 8);
        if (m == 0) atomicAdd(&lds_a[s * 16 + r + (hi16 << 3)], g);
      }
    }
  }
  __syncthreads();
  if (t < MT_ && n0 + t < N_)
    Alog[(size_t)b * N_ + n0 + t] = lds_a[t] + b_att[0];
}

// ---------- kernel 2: per-batch softmax stats (max, sum exp) ----------
__global__ __launch_bounds__(256) void k_stats(const float* __restrict__ Alog,
                                               float* __restrict__ stats) {
  const int b = blockIdx.x;
  const int t = threadIdx.x;
  const float* a = Alog + (size_t)b * N_;
  __shared__ float red[8];
  __shared__ float s_max;
  float mx = -3.4e38f;
  for (int i = t; i < N_; i += 256) mx = fmaxf(mx, a[i]);
  for (int off = 16; off; off >>= 1) mx = fmaxf(mx, __shfl_xor(mx, off));
  if ((t & 31) == 0) red[t >> 5] = mx;
  __syncthreads();
  if (t == 0) {
    float mm = red[0];
    for (int i = 1; i < 8; ++i) mm = fmaxf(mm, red[i]);
    s_max = mm;
  }
  __syncthreads();
  mx = s_max;
  float sum = 0.0f;
  for (int i = t; i < N_; i += 256) sum += expf(a[i] - mx);
  for (int off = 16; off; off >>= 1) sum += __shfl_xor(sum, off);
  if ((t & 31) == 0) red[t >> 5] = sum;
  __syncthreads();
  if (t == 0) {
    float s = 0.0f;
    for (int i = 0; i < 8; ++i) s += red[i];
    stats[b * 2]     = mx;
    stats[b * 2 + 1] = s;
  }
}

// ---------- kernel 3: softmax-weighted pooling ----------
__global__ __launch_bounds__(256) void k_wsum(const unsigned short* __restrict__ H,
                                              const float* __restrict__ Alog,
                                              const float* __restrict__ stats,
                                              float* __restrict__ M) {
  const int b  = blockIdx.y;
  const int ch = blockIdx.x;           // 80 chunks x 250 rows
  const int t  = threadIdx.x;
  const int n0 = ch * 250;
  const float mx  = stats[b * 2];
  const float inv = 1.0f / stats[b * 2 + 1];
  const float* ab = Alog + (size_t)b * N_;
  const unsigned short* Hb = H + (size_t)b * N_ * EMB_;
  float a0 = 0.0f, a1 = 0.0f;
  for (int n = n0; n < n0 + 250; ++n) {
    float wn = expf(ab[n] - mx) * inv;
    const unsigned short* hr = Hb + (size_t)n * EMB_;
    a0 += wn * bf2f(hr[t]);
    a1 += wn * bf2f(hr[t + 256]);
  }
  atomicAdd(&M[b * EMB_ + t], a0);
  atomicAdd(&M[b * EMB_ + t + 256], a1);
}

// ---------- kernel 4: tiny classifier ----------
__global__ __launch_bounds__(256) void k_cls(const float* __restrict__ M,
                                             const float* __restrict__ Wc,
                                             const float* __restrict__ bc,
                                             float* __restrict__ out) {
  const int t = threadIdx.x;
  const int g = t >> 5;
  const int lane = t & 31;
  const int b = g >> 1;
  const int c = g & 1;
  float s = 0.0f;
  for (int e = lane; e < EMB_; e += 32) s += M[b * EMB_ + e] * Wc[e * 2 + c];
  s += __shfl_xor(s, 1);
  s += __shfl_xor(s, 2);
  s += __shfl_xor(s, 4);
  s += __shfl_xor(s, 8);
  s += __shfl_xor(s, 16);
  if (lane == 0) out[b * 2 + c] = s + bc[c];
}

extern "C" void kernel_launch(void* const* d_in, const int* in_sizes, int n_in,
                              void* d_out, int out_size, void* d_ws, size_t ws_size,
                              hipStream_t stream) {
  (void)in_sizes; (void)n_in; (void)out_size; (void)ws_size;
  const float* x     = (const float*)d_in[0];
  const float* W_pe  = (const float*)d_in[1];
  const float* b_pe  = (const float*)d_in[2];
  const float* W_V   = (const float*)d_in[3];
  const float* b_V   = (const float*)d_in[4];
  const float* W_U   = (const float*)d_in[5];
  const float* b_U   = (const float*)d_in[6];
  const float* w_att = (const float*)d_in[7];
  const float* b_att = (const float*)d_in[8];
  const float* W_cls = (const float*)d_in[9];
  const float* b_cls = (const float*)d_in[10];

  char* ws = (char*)d_ws;
  unsigned*       WPEP  = (unsigned*)(ws + OFF_WPEP);
  unsigned*       WVP   = (unsigned*)(ws + OFF_WVP);
  unsigned*       WUP   = (unsigned*)(ws + OFF_WUP);
  unsigned short* H     = (unsigned short*)(ws + OFF_H);
  float*          Alog  = (float*)(ws + OFF_A);
  float*          stats = (float*)(ws + OFF_ST);
  float*          M     = (float*)(ws + OFF_M);

  k_prep <<<1288, 256, 0, stream>>>(W_pe, W_V, W_U, WPEP, WVP, WUP, M);
  k_main <<<dim3(NT1_, B_), 256, 0, stream>>>(x, b_pe, b_V, b_U, w_att, b_att,
                                              WPEP, WVP, WUP, H, Alog);
  k_stats<<<B_, 256, 0, stream>>>(Alog, stats);
  k_wsum <<<dim3(80, B_), 256, 0, stream>>>(H, Alog, stats, M);
  k_cls  <<<1, 256, 0, stream>>>(M, W_cls, b_cls, (float*)d_out);
}
